// Whitening2d_2388001817055
// MI455X (gfx1250) — matrix operand order repaired and checked by reading, not compiled, not run on hardware
//
#include <hip/hip_runtime.h>
#include <stdint.h>

// ---------------------------------------------------------------------------
// Whitening2d for MI455X (gfx1250, wave32, WMMA + async-LDS DMA).
//   x:[16,8192,256] f32 -> whitened output via Newton-Schulz inverse sqrt.
// Big GEMMs (K=8192 Gram, K=256 decorrelate): V_WMMA_F32_16X16X32_F16.
// Newton-Schulz 256^3 GEMMs: V_WMMA_F32_16X16X4_F32, operands staged with
// GLOBAL_LOAD_ASYNC_TO_LDS_B128, double-buffered via ASYNCcnt in-order rule.
// ---------------------------------------------------------------------------

typedef __attribute__((ext_vector_type(16))) _Float16 v16h;
typedef __attribute__((ext_vector_type(8)))  _Float16 v8h;
typedef __attribute__((ext_vector_type(2)))  _Float16 v2h;
typedef __attribute__((ext_vector_type(8)))  float    v8f;
typedef __attribute__((ext_vector_type(2)))  float    v2f;

constexpr int NB  = 16;    // batches
constexpr int SEQ = 8192;  // sequence length
constexpr int DIM = 256;   // feature dim
constexpr int KSL = 8;     // K-slices for Gram partials
constexpr int MCH = 32;    // chunks for mean partial reduction

// ----------------------- CDNA5 async LDS DMA helpers ------------------------
__device__ inline void async_load_b128(uint32_t lds_byte_addr, const void* g) {
  asm volatile("global_load_async_to_lds_b128 %0, %1, off"
               :: "v"(lds_byte_addr), "v"(g)
               : "memory");
}
__device__ inline void wait_asynccnt0() {
  asm volatile("s_wait_asynccnt 0x0" ::: "memory");
}
__device__ inline void wait_asynccnt2() {  // allow next-panel loads in flight
  asm volatile("s_wait_asynccnt 0x2" ::: "memory");
}
__device__ inline uint32_t lds_addr_of(const void* p) {
  // Generic (flat) LDS addresses: low 32 bits are the LDS offset (ISA 10.2).
  return (uint32_t)(uintptr_t)p;
}

// --------------------------- WMMA fragment gathers --------------------------
// A-matrix 16x32 f16 (ISA 7.12.2): lane L holds M = L%16; half h -> K =
// 16*(h/8) + 8*(L/16) + (h%8): two contiguous 16B LDS chunks.
__device__ inline v16h frag_a_f16(const _Float16* rowbase, int lane) {
  union { v16h v; v8h h[2]; } u;
  int koff = (lane >> 4) * 8;
  u.h[0] = *(const v8h*)(rowbase + koff);
  u.h[1] = *(const v8h*)(rowbase + 16 + koff);
  return u.v;
}
// B-matrix 32x16 f16: lane L holds N = L%16; half h -> K =
// 16*(h/8) + 8*(L/16) + (h%8) (same order as A).
__device__ inline v16h frag_b_f16(const _Float16* colbase, int lane) {
  union { v16h v; v8h h[2]; } u;
  int kb = (lane >> 4) * 8;
  u.h[0] = *(const v8h*)(colbase + kb);
  u.h[1] = *(const v8h*)(colbase + kb + 16);
  return u.v;
}

// ------------------------------- mean ---------------------------------------
__global__ __launch_bounds__(256) void k_mean_partial(
    const float* __restrict__ x, float* __restrict__ part) {
  int b = blockIdx.x / MCH, c = blockIdx.x % MCH;
  int d = threadIdx.x;
  const int rows = SEQ / MCH;  // 256
  const float* p = x + ((size_t)b * SEQ + (size_t)c * rows) * DIM + d;
  float s = 0.f;
  for (int i = 0; i < rows; ++i) s += p[(size_t)i * DIM];
  part[((size_t)b * MCH + c) * DIM + d] = s;
}

__global__ __launch_bounds__(256) void k_mean_final(
    const float* __restrict__ part, float* __restrict__ mean) {
  int idx = blockIdx.x * 256 + threadIdx.x;  // b*DIM + d
  int b = idx / DIM, d = idx % DIM;
  float s = 0.f;
  for (int c = 0; c < MCH; ++c) s += part[((size_t)b * MCH + c) * DIM + d];
  mean[idx] = s * (1.0f / SEQ);
}

// ------------------------- Gram = xn^T xn (K-sliced) -------------------------
// grid: (4 tiles [2x128 m x 2x128 n], KSL, NB); block 256 = 8 waves, each wave
// a 32x64 output region (2x4 f16 WMMA tiles).  Double-buffered LDS panels;
// staging = 2x2 f32 micro-tile (two b64 loads), register transpose, packed
// v2h (b32) LDS stores.
__device__ inline void stage_gram(const float* __restrict__ xrow0, int m0,
                                  int n0, const float* mA, const float* mB,
                                  _Float16* lA, _Float16* lB, int tid) {
  for (int t = 0; t < 4; ++t) {
    int u = tid + t * 256;
    int m = (u & 63) * 2, s = (u >> 6) * 2;
    const float* p = xrow0 + (size_t)s * DIM;
    {
      v2f r0 = *(const v2f*)(p + m0 + m);
      v2f r1 = *(const v2f*)(p + DIM + m0 + m);
      *(v2h*)&lA[m * 32 + s] =
          (v2h){(_Float16)(r0.x - mA[m]), (_Float16)(r1.x - mA[m])};
      *(v2h*)&lA[(m + 1) * 32 + s] =
          (v2h){(_Float16)(r0.y - mA[m + 1]), (_Float16)(r1.y - mA[m + 1])};
    }
    {
      v2f r0 = *(const v2f*)(p + n0 + m);
      v2f r1 = *(const v2f*)(p + DIM + n0 + m);
      *(v2h*)&lB[m * 32 + s] =
          (v2h){(_Float16)(r0.x - mB[m]), (_Float16)(r1.x - mB[m])};
      *(v2h*)&lB[(m + 1) * 32 + s] =
          (v2h){(_Float16)(r0.y - mB[m + 1]), (_Float16)(r1.y - mB[m + 1])};
    }
  }
}

__global__ __launch_bounds__(256) void k_gram(
    const float* __restrict__ x, const float* __restrict__ mean,
    float* __restrict__ gramp) {
  __shared__ _Float16 lA[2][128 * 32];  // m-major, k contiguous (2 x 8 KB)
  __shared__ _Float16 lB[2][128 * 32];  // n-major, k contiguous (2 x 8 KB)
  __shared__ float mA[128];
  __shared__ float mB[128];

  int b  = blockIdx.z;
  int ks = blockIdx.y;
  int m0 = (blockIdx.x & 1) * 128;
  int n0 = (blockIdx.x >> 1) * 128;
  int tid = threadIdx.x, lane = tid & 31, w = tid >> 5;
  int row0w = (w & 3) * 32, col0w = (w >> 2) * 64;

  if (tid < 128) {
    mA[tid] = mean[b * DIM + m0 + tid];
    mB[tid] = mean[b * DIM + n0 + tid];
  }
  __syncthreads();

  v8f acc[2][4] = {};
  const int KC = SEQ / KSL;  // 1024
  const int s0 = ks * KC;
  const float* xb = x + (size_t)b * SEQ * DIM;

  stage_gram(xb + (size_t)s0 * DIM, m0, n0, mA, mB, lA[0], lB[0], tid);
  __syncthreads();

  for (int kk = 0; kk < KC; kk += 32) {
    int cur = (kk >> 5) & 1;
    if (kk + 32 < KC)
      stage_gram(xb + (size_t)(s0 + kk + 32) * DIM, m0, n0, mA, mB,
                 lA[cur ^ 1], lB[cur ^ 1], tid);
    if (kk + 64 < KC) {  // near-cache prefetch two chunks ahead
      int s = tid >> 3, seg = tid & 7;
      __builtin_prefetch(
          &xb[(size_t)(s0 + kk + 64 + s) * DIM + ((tid & 128) ? n0 : m0) +
              seg * 16],
          0, 3);
    }
    for (int ti = 0; ti < 2; ++ti) {
      int m = row0w + ti * 16 + (lane & 15);
      v16h a = frag_a_f16(&lA[cur][m * 32], lane);
      for (int tj = 0; tj < 4; ++tj) {
        int n = col0w + tj * 16 + (lane & 15);
        v16h bb = frag_b_f16(&lB[cur][n * 32], lane);
        acc[ti][tj] = __builtin_amdgcn_wmma_f32_16x16x32_f16(
            false, a, false, bb, (short)0, acc[ti][tj], false, false);
      }
    }
    __syncthreads();
  }
  for (int ti = 0; ti < 2; ++ti)
    for (int tj = 0; tj < 4; ++tj)
      for (int r = 0; r < 8; ++r) {
        int row = m0 + row0w + ti * 16 + r + 8 * (lane >> 4);
        int col = n0 + col0w + tj * 16 + (lane & 15);
        gramp[(((size_t)b * KSL + ks) * DIM + row) * DIM + col] = acc[ti][tj][r];
      }
}

// ------------- sigma_n = (sum Gram)/((S-1)*tr) ; trace ; P = I --------------
__global__ __launch_bounds__(256) void k_sigma(
    const float* __restrict__ gramp, float* __restrict__ sign,
    float* __restrict__ Pm, float* __restrict__ trbuf,
    float* __restrict__ rtrbuf) {
  __shared__ float red[256];
  int b = blockIdx.x, t = threadIdx.x;
  float g = 0.f;
  for (int ks = 0; ks < KSL; ++ks)
    g += gramp[(((size_t)b * KSL + ks) * DIM + t) * DIM + t];
  red[t] = g / (float)(SEQ - 1);
  __syncthreads();
  for (int s = 128; s > 0; s >>= 1) {
    if (t < s) red[t] += red[t + s];
    __syncthreads();
  }
  float tr = red[0];
  if (t == 0) { trbuf[b] = tr; rtrbuf[b] = rsqrtf(tr); }
  float inv = 1.0f / ((float)(SEQ - 1) * tr);
  for (int i = 0; i < DIM; ++i) {
    size_t idx = (size_t)i * DIM + t;
    float gg = 0.f;
    for (int ks = 0; ks < KSL; ++ks)
      gg += gramp[((size_t)b * KSL + ks) * DIM * DIM + idx];
    sign[(size_t)b * DIM * DIM + idx] = gg * inv;
    Pm[(size_t)b * DIM * DIM + idx] = (i == t) ? 1.0f : 0.0f;
  }
}

// ---------------- batched 256x256x256 f32 GEMM via WMMA 16x16x4 -------------
// B operand must be SYMMETRIC (true for P and sigma_n): the n-major
// k-contiguous LDS panel is a straight contiguous copy -> async B128 DMA.
// Double-buffered: next panel's DMA is in flight during current compute;
// s_wait_asynccnt 2 relies on in-order async-load completion (ISA 4.1).
// mode 0: C = A*B     mode 1: C = 1.5*C - 0.5*(A*B)   (Newton-Schulz update)
__global__ __launch_bounds__(256) void k_gemm256(
    const float* __restrict__ A, const float* __restrict__ Bm,
    float* __restrict__ C, int mode) {
  __shared__ float lA[2][64 * 64];  // m-major, k contiguous (2 x 16 KB)
  __shared__ float lB[2][64 * 64];  // n-major, k contiguous (2 x 16 KB)
  int b = blockIdx.y;
  int m0 = (blockIdx.x & 3) * 64, n0 = (blockIdx.x >> 2) * 64;
  int tid = threadIdx.x, lane = tid & 31, w = tid >> 5;
  int row0w = (w & 3) * 16, col0w = (w >> 2) * 32;
  const float* Ab = A + (size_t)b * DIM * DIM;
  const float* Bb = Bm + (size_t)b * DIM * DIM;
  int r = tid >> 2, c = (tid & 3) * 4;  // 64 rows x 4 chunks of 4 floats
  uint32_t ldsoff = (uint32_t)(r * 64 + c) * 4;

  auto issue = [&](int buf, int kc) {
    async_load_b128(lds_addr_of(lA[buf]) + ldsoff,
                    Ab + (size_t)(m0 + r) * DIM + kc + c);
    async_load_b128(lds_addr_of(lB[buf]) + ldsoff,
                    Bb + (size_t)(n0 + r) * DIM + kc + c);  // B symmetric
  };

  v8f acc[2] = {};
  issue(0, 0);
  for (int kc = 0; kc < DIM; kc += 64) {
    int cur = (kc >> 6) & 1;
    if (kc + 64 < DIM) { issue(cur ^ 1, kc + 64); wait_asynccnt2(); }
    else               { wait_asynccnt0(); }
    __syncthreads();
    for (int kk = 0; kk < 64; kk += 4) {
      // f32 A 16x4: lane holds M=L%16, K = kk + 2*(L/16) + {0,1}
      int koff = kk + 2 * (lane >> 4);
      v2f a = *(const v2f*)&lA[cur][(row0w + (lane & 15)) * 64 + koff];
      for (int tj = 0; tj < 2; ++tj) {
        v2f bb =
            *(const v2f*)&lB[cur][(col0w + tj * 16 + (lane & 15)) * 64 + koff];
        acc[tj] = __builtin_amdgcn_wmma_f32_16x16x4_f32(
            false, a, false, bb, (short)0, acc[tj], false, false);
      }
    }
    __syncthreads();
  }
  for (int tj = 0; tj < 2; ++tj)
    for (int rr = 0; rr < 8; ++rr) {
      int row = m0 + row0w + rr + 8 * (lane >> 4);
      int col = n0 + col0w + tj * 16 + (lane & 15);
      size_t o = (size_t)b * DIM * DIM + (size_t)row * DIM + col;
      float v = acc[tj][rr];
      C[o] = (mode == 0) ? v : (1.5f * C[o] - 0.5f * v);
    }
}

// ---- wmT = (P*rsqrt(tr))^T (f16, e-major); cvec = (m @ P)*rsqrt(tr) --------
__global__ __launch_bounds__(256) void k_wmprep(
    const float* __restrict__ Pm, const float* __restrict__ mean,
    const float* __restrict__ rtrbuf, _Float16* __restrict__ wmhT,
    float* __restrict__ cvec) {
  __shared__ float mld[256];
  int b = blockIdx.x, t = threadIdx.x;
  mld[t] = mean[b * DIM + t];
  __syncthreads();
  float rtr = rtrbuf[b];
  const float* Pb = Pm + (size_t)b * DIM * DIM;
  float cv = 0.f;
  for (int d = 0; d < DIM; ++d) cv += mld[d] * Pb[(size_t)d * DIM + t];
  cvec[b * DIM + t] = cv * rtr;
  _Float16* wb = wmhT + (size_t)b * DIM * DIM;
  for (int i = 0; i < DIM; ++i) {
    // transposed store: wmhT[e=t][d=i]  (k-contiguous rows for async staging)
    wb[(size_t)t * DIM + i] = (_Float16)(Pb[(size_t)i * DIM + t] * rtr);
  }
}

// ------------------- out = x @ wm - cvec  (f16 WMMA, K=256) -----------------
// grid: (SEQ/128, DIM/64, NB); block 256 = 8 waves, wave = 32x32 region.
// B panel (wmhT rows n0..n0+63, all K) = one contiguous 32KB block -> async.
// A panel double-buffered, b64 global loads + packed v2h LDS stores.
__device__ inline void stage_decor(const float* __restrict__ xbase,
                                   _Float16* lA, int tid) {
  for (int i = 0; i < 8; ++i) {
    int u = tid + i * 256;
    int d = (u & 15) * 2, s = u >> 4;  // 16 d-pairs x 128 s
    v2f r = *(const v2f*)(xbase + (size_t)s * DIM + d);
    *(v2h*)&lA[s * 32 + d] = (v2h){(_Float16)r.x, (_Float16)r.y};
  }
}

__global__ __launch_bounds__(256) void k_decor(
    const float* __restrict__ x, const _Float16* __restrict__ wmhT,
    const float* __restrict__ cvec, float* __restrict__ out) {
  __shared__ _Float16 lB[64 * 256];     // n-major, k contiguous (32 KB)
  __shared__ _Float16 lA[2][128 * 32];  // s-major, k contiguous (2 x 8 KB)
  int b  = blockIdx.z;
  int s0 = blockIdx.x * 128;
  int n0 = blockIdx.y * 64;
  int tid = threadIdx.x, lane = tid & 31, w = tid >> 5;
  int row0w = (w & 3) * 32, col0w = (w >> 2) * 32;
  const float* xbase = x + ((size_t)b * SEQ + s0) * DIM;

  // async DMA the whole wm panel: contiguous global -> contiguous LDS
  {
    const _Float16* src = wmhT + (size_t)b * DIM * DIM + (size_t)n0 * DIM;
    uint32_t dst = lds_addr_of(lB);
    for (int i = 0; i < 8; ++i) {
      int idx = tid + i * 256;  // 2048 x 16B = 32KB
      async_load_b128(dst + (uint32_t)idx * 16, (const char*)src + idx * 16);
    }
  }
  stage_decor(xbase, lA[0], tid);
  wait_asynccnt0();
  __syncthreads();

  v8f acc[2][2] = {};
  for (int kc = 0; kc < DIM; kc += 32) {
    int cur = (kc >> 5) & 1;
    if (kc + 32 < DIM) stage_decor(xbase + kc + 32, lA[cur ^ 1], tid);
    for (int ti = 0; ti < 2; ++ti) {
      int m = row0w + ti * 16 + (lane & 15);
      v16h a = frag_a_f16(&lA[cur][m * 32], lane);
      for (int tj = 0; tj < 2; ++tj) {
        int n = col0w + tj * 16 + (lane & 15);
        v16h bb = frag_b_f16(&lB[n * 256 + kc], lane);
        acc[ti][tj] = __builtin_amdgcn_wmma_f32_16x16x32_f16(
            false, a, false, bb, (short)0, acc[ti][tj], false, false);
      }
    }
    __syncthreads();
  }
  for (int ti = 0; ti < 2; ++ti)
    for (int tj = 0; tj < 2; ++tj)
      for (int r = 0; r < 8; ++r) {
        int row = s0 + row0w + ti * 16 + r + 8 * (lane >> 4);
        int col = n0 + col0w + tj * 16 + (lane & 15);
        out[((size_t)b * SEQ + row) * DIM + col] =
            acc[ti][tj][r] - cvec[b * DIM + col];
      }
}

// ---------------------------------------------------------------------------
extern "C" void kernel_launch(void* const* d_in, const int* in_sizes, int n_in,
                              void* d_out, int out_size, void* d_ws,
                              size_t ws_size, hipStream_t stream) {
  (void)in_sizes; (void)n_in; (void)out_size; (void)ws_size;
  const float* x = (const float*)d_in[0];
  float* out = (float*)d_out;

  char* ws = (char*)d_ws;
  size_t off = 0;
  auto alloc = [&](size_t bytes) -> void* {
    void* p = ws + off;
    off += (bytes + 255) & ~(size_t)255;
    return p;
  };
  float*    mpart = (float*)alloc((size_t)NB * MCH * DIM * 4);
  float*    mean  = (float*)alloc((size_t)NB * DIM * 4);
  float*    trb   = (float*)alloc((size_t)NB * 4);
  float*    rtrb  = (float*)alloc((size_t)NB * 4);
  float*    gramp = (float*)alloc((size_t)NB * KSL * DIM * DIM * 4); // 33.5 MB
  float*    sign  = (float*)alloc((size_t)NB * DIM * DIM * 4);       // 4 MB
  float*    Pm    = (float*)alloc((size_t)NB * DIM * DIM * 4);       // 4 MB
  float*    T1    = (float*)alloc((size_t)NB * DIM * DIM * 4);       // 4 MB
  float*    T2    = (float*)alloc((size_t)NB * DIM * DIM * 4);       // 4 MB
  _Float16* wmhT  = (_Float16*)alloc((size_t)NB * DIM * DIM * 2);    // 2 MB
  float*    cvec  = (float*)alloc((size_t)NB * DIM * 4);

  // 1) mean over sequence axis
  k_mean_partial<<<NB * MCH, 256, 0, stream>>>(x, mpart);
  k_mean_final<<<NB, 256, 0, stream>>>(mpart, mean);

  // 2) Gram of centered data (f16 WMMA, K split 8 ways)
  k_gram<<<dim3(4, KSL, NB), 256, 0, stream>>>(x, mean, gramp);

  // 3) sigma_n, trace, P = I
  k_sigma<<<NB, 256, 0, stream>>>(gramp, sign, Pm, trb, rtrb);

  // 4) Newton-Schulz: P = 1.5P - 0.5*(P*P*P*sigma_n), 4 iterations (f32 WMMA)
  for (int it = 0; it < 4; ++it) {
    k_gemm256<<<dim3(16, NB), 256, 0, stream>>>(Pm, Pm, T1, 0);
    k_gemm256<<<dim3(16, NB), 256, 0, stream>>>(T1, Pm, T2, 0);
    k_gemm256<<<dim3(16, NB), 256, 0, stream>>>(T2, sign, Pm, 1);
  }

  // 5) wm^T = (P/sqrt(tr))^T as f16, mean folded into cvec
  k_wmprep<<<NB, 256, 0, stream>>>(Pm, mean, rtrb, wmhT, cvec);

  // 6) out = x @ wm - cvec  (f16 WMMA, async-DMA'd B panel)
  k_decor<<<dim3(SEQ / 128, DIM / 64, NB), 256, 0, stream>>>(x, wmhT, cvec, out);
}
